// MutationGraphiT_54185307407185
// MI455X (gfx1250) — matrix operand order repaired and checked by reading, not compiled, not run on hardware
//
#include <hip/hip_runtime.h>
#include <hip/hip_bf16.h>

#define BB    8
#define FF    3
#define NN    1024
#define VOC   32
#define EE    64
#define DD    256
#define HH    8
#define HDIM  32
#define LL    4
#define CC    36
#define THRS  0.6f
#define ROWS  (BB*NN)

typedef _Float16 half_t;
typedef _Float16 v16h __attribute__((ext_vector_type(16)));
typedef float    v8f  __attribute__((ext_vector_type(8)));

__device__ __forceinline__ v8f wmma_f16(v16h a, v16h b, v8f c) {
  // D = A(16x32 f16) x B(32x16 f16) + C(16x16 f32)
  return __builtin_amdgcn_wmma_f32_16x16x32_f16(false, a, false, b, (short)0, c, false, false);
}

// ---------------------------------------------------------------------------
// Generic WMMA GEMM:  C[M,Nc] = A[M,K](f16) * W[Nc,K](f16)^T + bias, opt ReLU.
// One wave computes a 32x32 tile (2x2 of 16x16 WMMA tiles).
// ---------------------------------------------------------------------------
__global__ void gemm_wmma_kernel(const half_t* __restrict__ A,
                                 const half_t* __restrict__ W,
                                 const float*  __restrict__ bias,
                                 float*  __restrict__ C,
                                 half_t* __restrict__ C16,
                                 int M, int Nc, int K, int do_relu)
{
  const int lane = threadIdx.x & 31;
  const int hf   = lane >> 4;      // half-wave id
  const int r    = lane & 15;
  const int m0 = blockIdx.x * 32;
  const int n0 = blockIdx.y * 32;

  v8f acc00 = {}, acc01 = {}, acc10 = {}, acc11 = {};

  const half_t* Ar0 = A + (size_t)(m0 + r)      * K;
  const half_t* Ar1 = A + (size_t)(m0 + 16 + r) * K;
  const half_t* Wr0 = W + (size_t)(n0 + r)      * K;
  const half_t* Wr1 = W + (size_t)(n0 + 16 + r) * K;

  for (int k0 = 0; k0 < K; k0 += 32) {
    v16h a0, a1, b0, b1;
#pragma unroll
    for (int i = 0; i < 8; ++i) {
      // A layout (16-bit 16x32): lane half hf -> K = 8*hf+i (v[0..7]), 16+8*hf+i (v[8..15])
      a0[i]     = Ar0[k0 + 8*hf + i];
      a0[i + 8] = Ar0[k0 + 16 + 8*hf + i];
      a1[i]     = Ar1[k0 + 8*hf + i];
      a1[i + 8] = Ar1[k0 + 16 + 8*hf + i];
    }
#pragma unroll
    for (int i = 0; i < 8; ++i) {
      // B layout (16-bit 32x16): col = lane&15, K = 8*hf+i (v[0..7]), 16+8*hf+i (v[8..15]).  B[k][n] = W[n][k].
      b0[i] = Wr0[k0 + 8*hf + i];  b0[i + 8] = Wr0[k0 + 16 + 8*hf + i];
      b1[i] = Wr1[k0 + 8*hf + i];  b1[i + 8] = Wr1[k0 + 16 + 8*hf + i];
    }
    acc00 = wmma_f16(a0, b0, acc00);
    acc01 = wmma_f16(a0, b1, acc01);
    acc10 = wmma_f16(a1, b0, acc10);
    acc11 = wmma_f16(a1, b1, acc11);
  }

#pragma unroll
  for (int reg = 0; reg < 8; ++reg) {
    const int row0 = m0 + reg + 8*hf;   // C/D layout: M = reg + 8*half
    const int row1 = row0 + 16;
    const int col0 = n0 + r;
    const int col1 = col0 + 16;
    float v00 = acc00[reg], v01 = acc01[reg], v10 = acc10[reg], v11 = acc11[reg];
    if (bias) { v00 += bias[col0]; v01 += bias[col1]; v10 += bias[col0]; v11 += bias[col1]; }
    if (do_relu) {
      v00 = fmaxf(v00, 0.f); v01 = fmaxf(v01, 0.f);
      v10 = fmaxf(v10, 0.f); v11 = fmaxf(v11, 0.f);
    }
    if (C) {
      C[(size_t)row0*Nc + col0] = v00;  C[(size_t)row0*Nc + col1] = v01;
      C[(size_t)row1*Nc + col0] = v10;  C[(size_t)row1*Nc + col1] = v11;
    }
    if (C16) {
      C16[(size_t)row0*Nc + col0] = (half_t)v00;  C16[(size_t)row0*Nc + col1] = (half_t)v01;
      C16[(size_t)row1*Nc + col0] = (half_t)v10;  C16[(size_t)row1*Nc + col1] = (half_t)v11;
    }
  }
}

// ---------------------------------------------------------------------------
// Flash attention: one wave per (b, h, 16-row tile). qkv16: [B*N, 3D] f16,
// q at +0, k at +D, v at +2D, head h occupies columns h*32..h*32+31.
// ---------------------------------------------------------------------------
__global__ void flash_attn_kernel(const half_t* __restrict__ qkv,
                                  half_t* __restrict__ o16)
{
  __shared__ half_t ps[16 * 32];
  const int lane = threadIdx.x & 31;
  const int hf = lane >> 4, r = lane & 15;
  const int b = blockIdx.x / HH, h = blockIdx.x % HH;
  const int m0 = blockIdx.y * 16;
  const int LD = 3 * DD;

  const half_t* Qp = qkv + (size_t)b * NN * LD + h * HDIM;
  const half_t* Kp = Qp + DD;
  const half_t* Vp = Qp + 2 * DD;

  v16h qa;
  {
    const half_t* qrow = Qp + (size_t)(m0 + r) * LD;
#pragma unroll
    for (int i = 0; i < 8; ++i) { qa[i] = qrow[8*hf + i]; qa[i+8] = qrow[16 + 8*hf + i]; }
  }

  v8f o0 = {}, o1 = {};
  float mrun[8], lrun[8];
#pragma unroll
  for (int g = 0; g < 8; ++g) { mrun[g] = -3.0e38f; lrun[g] = 0.f; }
  const float scale = 0.17677669529663687f;   // 1/sqrt(32)

  for (int j0 = 0; j0 < NN; j0 += 32) {
    v16h bk0, bk1;
    const half_t* k0p = Kp + (size_t)(j0 + r)      * LD;
    const half_t* k1p = Kp + (size_t)(j0 + 16 + r) * LD;
#pragma unroll
    for (int i = 0; i < 16; ++i) { const int kk = (i < 8) ? 8*hf + i : 16 + 8*hf + (i - 8); bk0[i] = k0p[kk]; bk1[i] = k1p[kk]; }

    v8f z = {};
    v8f s0 = wmma_f16(qa, bk0, z);
    v8f s1 = wmma_f16(qa, bk1, z);

    float p0[8], p1[8];
#pragma unroll
    for (int g = 0; g < 8; ++g) {
      float a = s0[g] * scale, c = s1[g] * scale;
      float mx = fmaxf(a, c);
      for (int m = 8; m >= 1; m >>= 1) mx = fmaxf(mx, __shfl_xor(mx, m, 32)); // row lives in one half-wave
      float mnew  = fmaxf(mrun[g], mx);
      float alpha = __expf(mrun[g] - mnew);
      float e0 = __expf(a - mnew), e1 = __expf(c - mnew);
      float rs = e0 + e1;
      for (int m = 8; m >= 1; m >>= 1) rs += __shfl_xor(rs, m, 32);
      lrun[g] = lrun[g] * alpha + rs;
      mrun[g] = mnew;
      o0[g] *= alpha; o1[g] *= alpha;
      p0[g] = e0; p1[g] = e1;
    }

    __syncthreads();
#pragma unroll
    for (int g = 0; g < 8; ++g) {            // C-layout -> LDS [16][32]
      ps[(g + 8*hf)*32 + r]      = (half_t)p0[g];
      ps[(g + 8*hf)*32 + 16 + r] = (half_t)p1[g];
    }
    __syncthreads();

    v16h pa;
#pragma unroll
    for (int i = 0; i < 8; ++i) {            // LDS -> A-layout
      pa[i]     = ps[r*32 + 8*hf + i];
      pa[i + 8] = ps[r*32 + 16 + 8*hf + i];
    }
    v16h bv0, bv1;
#pragma unroll
    for (int i = 0; i < 16; ++i) {
      const half_t* vrow = Vp + (size_t)(j0 + ((i < 8) ? 8*hf + i : 16 + 8*hf + (i - 8))) * LD;
      bv0[i] = vrow[r];
      bv1[i] = vrow[16 + r];
    }
    o0 = wmma_f16(pa, bv0, o0);
    o1 = wmma_f16(pa, bv1, o1);
  }

#pragma unroll
  for (int g = 0; g < 8; ++g) {
    const int row = m0 + g + 8*hf;
    const float inv = 1.0f / lrun[g];
    const size_t base = ((size_t)b * NN + row) * DD + h * HDIM;
    o16[base + r]      = (half_t)(o0[g] * inv);
    o16[base + 16 + r] = (half_t)(o1[g] * inv);
  }
}

// ---------------------------------------------------------------------------
// Embedding gather + count linear + positional encoding.
// ---------------------------------------------------------------------------
__global__ void embed_kernel(const int* __restrict__ cat, const float* __restrict__ cnt,
                             const float* __restrict__ emb, const float* __restrict__ cw,
                             const float* __restrict__ cb,  const float* __restrict__ pos,
                             float* __restrict__ x, half_t* __restrict__ x16)
{
  const int row = blockIdx.x, b = row / NN, n = row % NN, c = threadIdx.x;
  const int f = c >> 6, e = c & 63;
  float v;
  if (f < FF) {
    const int idx = cat[((size_t)b*FF + f)*NN + n];
    v = emb[((size_t)f*VOC + idx)*EE + e];
  } else {
    v = cnt[(size_t)b*NN + n] * cw[e] + cb[e];
  }
  v += pos[(size_t)n*DD + c];
  x  [(size_t)row*DD + c] = v;
  x16[(size_t)row*DD + c] = (half_t)v;
}

// ---------------------------------------------------------------------------
// Band adjacency mask [B*N, 9] (offsets -4..4): mask>0 per reference rules.
// ---------------------------------------------------------------------------
__global__ void adj_kernel(const int* __restrict__ cat, const float* __restrict__ cnt,
                           float* __restrict__ band)
{
  const int t = blockIdx.x * blockDim.x + threadIdx.x;
  if (t >= ROWS) return;
  const int b = t / NN, i = t % NN;
#pragma unroll
  for (int dd = 0; dd < 9; ++dd) {
    const int off = dd - 4;
    const int j = i + off;
    float m = 0.f;
    if (j >= 0 && j < NN) {
      const int ao = off < 0 ? -off : off;
      if (ao <= 1) {
        m = 1.f;                     // diag & k=+-1 always positive
      } else {
        float cs = 0.f;
        for (int f = 0; f < FF; ++f)
          cs += (cat[((size_t)b*FF + f)*NN + i] == cat[((size_t)b*FF + f)*NN + j]) ? 1.f : 0.f;
        cs *= (1.f / 3.f);
        const float ns = __expf(-fabsf(cnt[(size_t)b*NN + i] - cnt[(size_t)b*NN + j]) * 0.01f);
        const float sim = 0.5f * (cs + ns);
        m = (sim > THRS) ? 1.f : 0.f;
      }
    }
    band[(size_t)t * 9 + dd] = m;
  }
}

// GAT scalar terms: e_src = Wh . a[:D], e_dst = Wh . a[D:], one wave per row.
__global__ void gat_e_kernel(const float* __restrict__ Wh, const float* __restrict__ a,
                             float* __restrict__ es, float* __restrict__ ed)
{
  const int row = blockIdx.x, lane = threadIdx.x;
  const float* w = Wh + (size_t)row * DD;
  float s = 0.f, d = 0.f;
  for (int c = lane; c < DD; c += 32) { const float v = w[c]; s += v * a[c]; d += v * a[DD + c]; }
  for (int m = 16; m >= 1; m >>= 1) { s += __shfl_xor(s, m, 32); d += __shfl_xor(d, m, 32); }
  if (lane == 0) { es[row] = s; ed[row] = d; }
}

// Banded softmax aggregate + ELU, one wave per row.
__global__ void gat_agg_kernel(const float* __restrict__ Wh, const float* __restrict__ band,
                               const float* __restrict__ es, const float* __restrict__ ed,
                               float* __restrict__ out)
{
  const int row = blockIdx.x, b = row / NN, i = row % NN, lane = threadIdx.x;
  float lg[9]; bool ok[9];
  float mx = -3.0e38f;
#pragma unroll
  for (int dd = 0; dd < 9; ++dd) {
    const int j = i + dd - 4;
    const bool v = (j >= 0) && (j < NN) && (band[(size_t)row*9 + dd] > 0.f);
    ok[dd] = v;
    lg[dd] = v ? es[row] + ed[(size_t)b*NN + j] : -3.0e38f;
    mx = fmaxf(mx, lg[dd]);
  }
  float p[9]; float den = 0.f;
#pragma unroll
  for (int dd = 0; dd < 9; ++dd) { p[dd] = ok[dd] ? __expf(lg[dd] - mx) : 0.f; den += p[dd]; }
  const float inv = 1.f / den;
  for (int c = lane; c < DD; c += 32) {
    float acc = 0.f;
#pragma unroll
    for (int dd = 0; dd < 9; ++dd)
      if (ok[dd]) acc += p[dd] * Wh[((size_t)b*NN + (i + dd - 4)) * DD + c];
    acc *= inv;
    out[(size_t)row*DD + c] = acc > 0.f ? acc : (__expf(acc) - 1.f);   // ELU
  }
}

// Fused residual + LayerNorm, writes f32 master and f16 mirror. One wave/row.
__global__ void residual_ln_kernel(float* __restrict__ x, half_t* __restrict__ x16,
                                   const float* __restrict__ t,
                                   const float* __restrict__ g, const float* __restrict__ be)
{
  const int row = blockIdx.x, lane = threadIdx.x;
  float v[8]; float s = 0.f, sq = 0.f;
#pragma unroll
  for (int k = 0; k < 8; ++k) {
    const int c = lane + 32*k;
    const float y = x[(size_t)row*DD + c] + t[(size_t)row*DD + c];
    v[k] = y; s += y; sq += y * y;
  }
  for (int m = 16; m >= 1; m >>= 1) { s += __shfl_xor(s, m, 32); sq += __shfl_xor(sq, m, 32); }
  const float mean = s * (1.f / DD);
  const float var  = sq * (1.f / DD) - mean * mean;
  const float inv  = rsqrtf(var + 1e-5f);
#pragma unroll
  for (int k = 0; k < 8; ++k) {
    const int c = lane + 32*k;
    const float y = (v[k] - mean) * inv * g[c] + be[c];
    x  [(size_t)row*DD + c] = y;
    x16[(size_t)row*DD + c] = (half_t)y;
  }
}

__global__ void meanpool_kernel(const float* __restrict__ x, float* __restrict__ p)
{
  const int b = blockIdx.x, c = threadIdx.x;
  float s = 0.f;
  for (int n = 0; n < NN; ++n) s += x[((size_t)b*NN + n)*DD + c];
  p[b*DD + c] = s * (1.f / NN);
}

__global__ void classifier_kernel(const float* __restrict__ p,
                                  const float* __restrict__ w1, const float* __restrict__ b1,
                                  const float* __restrict__ w2, const float* __restrict__ b2,
                                  float* __restrict__ out)
{
  __shared__ float hid[DD/2];
  const int b = blockIdx.x, t = threadIdx.x;
  float s = b1[t];
  for (int k = 0; k < DD; ++k) s += p[b*DD + k] * w1[t*DD + k];
  hid[t] = fmaxf(s, 0.f);
  __syncthreads();
  if (t < CC) {
    float o = b2[t];
    for (int k = 0; k < DD/2; ++k) o += hid[k] * w2[t*(DD/2) + k];
    out[b*CC + t] = o;
  }
}

__global__ void cvt_kernel(const float* __restrict__ s, half_t* __restrict__ d, int n)
{
  const int i = blockIdx.x * blockDim.x + threadIdx.x;
  if (i < n) d[i] = (half_t)s[i];
}

// dst[c*R + r] = src[r*Cc + c]  (transpose, for gat_W used as x @ W)
__global__ void cvtT_kernel(const float* __restrict__ s, half_t* __restrict__ d, int R, int Cc)
{
  const int i = blockIdx.x * blockDim.x + threadIdx.x;
  if (i >= R * Cc) return;
  const int r = i / Cc, c = i % Cc;
  d[(size_t)c*R + r] = (half_t)s[(size_t)r*Cc + c];
}

// ---------------------------------------------------------------------------
extern "C" void kernel_launch(void* const* d_in, const int* in_sizes, int n_in,
                              void* d_out, int out_size, void* d_ws, size_t ws_size,
                              hipStream_t stream)
{
  (void)in_sizes; (void)n_in; (void)out_size; (void)ws_size;
  const int*   cat        = (const int*)  d_in[0];
  const float* cnt        = (const float*)d_in[1];
  const float* emb        = (const float*)d_in[2];
  const float* count_w    = (const float*)d_in[3];
  const float* count_b    = (const float*)d_in[4];
  const float* pos_enc    = (const float*)d_in[5];
  const float* gat_W      = (const float*)d_in[6];
  const float* gat_a      = (const float*)d_in[7];
  const float* attn_in_w  = (const float*)d_in[8];
  const float* attn_in_b  = (const float*)d_in[9];
  const float* attn_out_w = (const float*)d_in[10];
  const float* attn_out_b = (const float*)d_in[11];
  const float* lin1_w     = (const float*)d_in[12];
  const float* lin1_b     = (const float*)d_in[13];
  const float* lin2_w     = (const float*)d_in[14];
  const float* lin2_b     = (const float*)d_in[15];
  const float* ln_g       = (const float*)d_in[16];
  const float* ln_b       = (const float*)d_in[17];
  const float* cls1_w     = (const float*)d_in[18];
  const float* cls1_b     = (const float*)d_in[19];
  const float* cls2_w     = (const float*)d_in[20];
  const float* cls2_b     = (const float*)d_in[21];

  char* ws = (char*)d_ws;
  size_t off = 0;
  auto alloc = [&](size_t bytes) -> void* {
    void* p = ws + off;
    off = (off + bytes + 255) & ~(size_t)255;
    return p;
  };

  float*  x       = (float*) alloc((size_t)ROWS*DD*4);
  half_t* x16     = (half_t*)alloc((size_t)ROWS*DD*2);
  float*  band    = (float*) alloc((size_t)ROWS*9*4);
  half_t* gatWT16 = (half_t*)alloc((size_t)LL*DD*DD*2);
  half_t* inW16   = (half_t*)alloc((size_t)LL*3*DD*DD*2);
  half_t* outW16  = (half_t*)alloc((size_t)LL*DD*DD*2);
  half_t* l1W16   = (half_t*)alloc((size_t)LL*2*DD*DD*2);
  half_t* l2W16   = (half_t*)alloc((size_t)LL*2*DD*DD*2);
  float*  Wh      = (float*) alloc((size_t)ROWS*DD*4);
  half_t* buf16   = (half_t*)alloc((size_t)ROWS*3*DD*2);  // qkv16 / hidden16
  half_t* o16     = (half_t*)alloc((size_t)ROWS*DD*2);
  float*  buf2    = (float*) alloc((size_t)ROWS*DD*4);    // residual branch
  float*  es      = (float*) alloc((size_t)ROWS*4);
  float*  ed      = (float*) alloc((size_t)ROWS*4);
  float*  pooled  = (float*) alloc((size_t)BB*DD*4);

  // ---- one-time weight conversion to f16 ----
  const int CT = 256;
  cvt_kernel<<<(LL*3*DD*DD + CT-1)/CT, CT, 0, stream>>>(attn_in_w,  inW16,  LL*3*DD*DD);
  cvt_kernel<<<(LL*DD*DD   + CT-1)/CT, CT, 0, stream>>>(attn_out_w, outW16, LL*DD*DD);
  cvt_kernel<<<(LL*2*DD*DD + CT-1)/CT, CT, 0, stream>>>(lin1_w,     l1W16,  LL*2*DD*DD);
  cvt_kernel<<<(LL*2*DD*DD + CT-1)/CT, CT, 0, stream>>>(lin2_w,     l2W16,  LL*2*DD*DD);
  for (int l = 0; l < LL; ++l)
    cvtT_kernel<<<(DD*DD + CT-1)/CT, CT, 0, stream>>>(gat_W + (size_t)l*DD*DD,
                                                      gatWT16 + (size_t)l*DD*DD, DD, DD);

  // ---- input embedding + band adjacency ----
  embed_kernel<<<ROWS, DD, 0, stream>>>(cat, cnt, emb, count_w, count_b, pos_enc, x, x16);
  adj_kernel<<<(ROWS + CT-1)/CT, CT, 0, stream>>>(cat, cnt, band);

  // ---- transformer layers ----
  for (int l = 0; l < LL; ++l) {
    // GAT: Wh = x @ gat_W   (WMMA)
    gemm_wmma_kernel<<<dim3(ROWS/32, DD/32), 32, 0, stream>>>(
        x16, gatWT16 + (size_t)l*DD*DD, nullptr, Wh, nullptr, ROWS, DD, DD, 0);
    gat_e_kernel<<<ROWS, 32, 0, stream>>>(Wh, gat_a + (size_t)l*2*DD, es, ed);
    gat_agg_kernel<<<ROWS, 32, 0, stream>>>(Wh, band, es, ed, buf2);
    residual_ln_kernel<<<ROWS, 32, 0, stream>>>(x, x16, buf2,
        ln_g + (size_t)(l*3 + 0)*DD, ln_b + (size_t)(l*3 + 0)*DD);

    // MHA: QKV projection (WMMA, f16 out), flash attention (WMMA), out proj (WMMA)
    gemm_wmma_kernel<<<dim3(ROWS/32, (3*DD)/32), 32, 0, stream>>>(
        x16, inW16 + (size_t)l*3*DD*DD, attn_in_b + (size_t)l*3*DD,
        nullptr, buf16, ROWS, 3*DD, DD, 0);
    flash_attn_kernel<<<dim3(BB*HH, NN/16), 32, 0, stream>>>(buf16, o16);
    gemm_wmma_kernel<<<dim3(ROWS/32, DD/32), 32, 0, stream>>>(
        o16, outW16 + (size_t)l*DD*DD, attn_out_b + (size_t)l*DD,
        buf2, nullptr, ROWS, DD, DD, 0);
    residual_ln_kernel<<<ROWS, 32, 0, stream>>>(x, x16, buf2,
        ln_g + (size_t)(l*3 + 1)*DD, ln_b + (size_t)(l*3 + 1)*DD);

    // FFN: D -> 2D (ReLU, f16 out) -> D (WMMA both)
    gemm_wmma_kernel<<<dim3(ROWS/32, (2*DD)/32), 32, 0, stream>>>(
        x16, l1W16 + (size_t)l*2*DD*DD, lin1_b + (size_t)l*2*DD,
        nullptr, buf16, ROWS, 2*DD, DD, 1);
    gemm_wmma_kernel<<<dim3(ROWS/32, DD/32), 32, 0, stream>>>(
        buf16, l2W16 + (size_t)l*2*DD*DD, lin2_b + (size_t)l*DD,
        buf2, nullptr, ROWS, DD, 2*DD, 0);
    residual_ln_kernel<<<ROWS, 32, 0, stream>>>(x, x16, buf2,
        ln_g + (size_t)(l*3 + 2)*DD, ln_b + (size_t)(l*3 + 2)*DD);
  }

  // ---- pool + classifier ----
  meanpool_kernel<<<BB, DD, 0, stream>>>(x, pooled);
  classifier_kernel<<<BB, DD/2, 0, stream>>>(pooled, cls1_w, cls1_b, cls2_w, cls2_b,
                                             (float*)d_out);
}
